// LSTMSeq2Seq_57801669870211
// MI455X (gfx1250) — compile-verified, matrix operand order checked by reading, not run on hardware
//
#include <hip/hip_runtime.h>
#include <hip/hip_bf16.h>

typedef __bf16 bf16_t;
typedef bf16_t v16bf __attribute__((ext_vector_type(16)));
typedef float  v8f   __attribute__((ext_vector_type(8)));
typedef unsigned int v4u __attribute__((ext_vector_type(4)));

#define HID   512
#define GATES 2048
#define BATCH 1024
#define T_IN  336
#define T_OUT 168
#define FEAT  16

__device__ __forceinline__ unsigned short f2bf(float f) {
    unsigned u = __float_as_uint(f);
    u += 0x7fffu + ((u >> 16) & 1u);      // round to nearest even
    return (unsigned short)(u >> 16);
}
__device__ __forceinline__ float sigm(float x) { return 1.0f / (1.0f + __expf(-x)); }
__device__ __forceinline__ float tanh_fast(float x) {
    float e = __expf(2.0f * x);
    return 1.0f - 2.0f / (e + 1.0f);      // -> +/-1 at saturation, no NaN
}

union Frag { v16bf v; v4u q[2]; };

// 16-bit A/B 16x32 WMMA fragment (ISA 7.12.2):
// lanes 0-15 hold row, K = [k..k+7] and [k+16..k+23]; lanes 16-31: K=[k+8..k+15],[k+24..k+31].
__device__ __forceinline__ v16bf load_frag(const unsigned short* __restrict__ base,
                                           int row, int ld, int k) {
    const int lane = threadIdx.x & 31;
    const int r  = row + (lane & 15);
    const int k0 = k + ((lane & 16) ? 8 : 0);
    const unsigned short* p = base + (size_t)r * ld + k0;
    Frag u;
    u.q[0] = *(const v4u*)(p);
    u.q[1] = *(const v4u*)(p + 16);
    return u.v;
}

// K2 == 16: only K 0..15 valid, zero-pad upper half of the fragment.
__device__ __forceinline__ v16bf load_frag_k16(const unsigned short* __restrict__ base,
                                               int row, int ld) {
    const int lane = threadIdx.x & 31;
    const int r  = row + (lane & 15);
    const int k0 = (lane & 16) ? 8 : 0;
    const v4u z = {0u, 0u, 0u, 0u};
    Frag u;
    u.q[0] = *(const v4u*)(base + (size_t)r * ld + k0);
    u.q[1] = z;
    return u.v;
}

// One LSTM cell step, fused GEMM + gates.
// gates[B,4H] = Aprev[B,H] @ Whh^T + A2[B,K2] @ Wih^T (+ r1x outer r1w) + bias
// Block: 64 rows x 64 hidden cols, all 4 gate groups -> full cell update in epilogue.
__global__ __launch_bounds__(256, 1)
void lstm_cell_kernel(const unsigned short* __restrict__ Aprev,   // [B,H] bf16
                      const unsigned short* __restrict__ A2,      // [B,K2] bf16 or null
                      int K2, int lda2,
                      const unsigned short* __restrict__ Whh,     // [4H,H] bf16
                      const unsigned short* __restrict__ Wih,     // [4H,K2] bf16 or null
                      const float* __restrict__ r1x,              // [B] or null (decoder scalar in)
                      const float* __restrict__ r1w,              // [4H] or null
                      const float* __restrict__ bias,             // [4H]
                      float* __restrict__ C,                      // [B,H] f32 in/out
                      unsigned short* __restrict__ Hout)          // [B,H] bf16
{
    const int lane = threadIdx.x & 31;
    const int wave = threadIdx.x >> 5;
    const int wm = wave & 1;                       // 2 row groups of 32
    const int wj = wave >> 1;                      // 4 hidden-col tiles of 16
    const int m0 = blockIdx.x * 64 + wm * 32;      // wave's first M tile
    const int j0 = blockIdx.y * 64 + wj * 16;      // hidden column tile

    const v8f vz = {0.f, 0.f, 0.f, 0.f, 0.f, 0.f, 0.f, 0.f};
    v8f acc[2][4];
#pragma unroll
    for (int mt = 0; mt < 2; ++mt)
#pragma unroll
        for (int g = 0; g < 4; ++g) acc[mt][g] = vz;

    // Recurrent part: K = 512
    for (int k = 0; k < HID; k += 32) {
        v16bf a0 = load_frag(Aprev, m0,      HID, k);
        v16bf a1 = load_frag(Aprev, m0 + 16, HID, k);
#pragma unroll
        for (int g = 0; g < 4; ++g) {
            v16bf b = load_frag(Whh, g * HID + j0, HID, k);
            acc[0][g] = __builtin_amdgcn_wmma_f32_16x16x32_bf16(false, a0, false, b,
                                                                (short)0, acc[0][g], false, false);
            acc[1][g] = __builtin_amdgcn_wmma_f32_16x16x32_bf16(false, a1, false, b,
                                                                (short)0, acc[1][g], false, false);
        }
    }
    // Input part
    if (K2 == HID) {
        for (int k = 0; k < HID; k += 32) {
            v16bf a0 = load_frag(A2, m0,      lda2, k);
            v16bf a1 = load_frag(A2, m0 + 16, lda2, k);
#pragma unroll
            for (int g = 0; g < 4; ++g) {
                v16bf b = load_frag(Wih, g * HID + j0, HID, k);
                acc[0][g] = __builtin_amdgcn_wmma_f32_16x16x32_bf16(false, a0, false, b,
                                                                    (short)0, acc[0][g], false, false);
                acc[1][g] = __builtin_amdgcn_wmma_f32_16x16x32_bf16(false, a1, false, b,
                                                                    (short)0, acc[1][g], false, false);
            }
        }
    } else if (K2 == FEAT) {
        v16bf a0 = load_frag_k16(A2, m0,      lda2);
        v16bf a1 = load_frag_k16(A2, m0 + 16, lda2);
#pragma unroll
        for (int g = 0; g < 4; ++g) {
            v16bf b = load_frag_k16(Wih, g * HID + j0, FEAT);
            acc[0][g] = __builtin_amdgcn_wmma_f32_16x16x32_bf16(false, a0, false, b,
                                                                (short)0, acc[0][g], false, false);
            acc[1][g] = __builtin_amdgcn_wmma_f32_16x16x32_bf16(false, a1, false, b,
                                                                (short)0, acc[1][g], false, false);
        }
    }

    // Epilogue: gate nonlinearities + cell update (C/D layout: VGPR r, lanes 0-15 -> M=r,
    // lanes 16-31 -> M=r+8; N = lane&15).
    const int col = j0 + (lane & 15);
    const float bi = bias[col];
    const float bf = bias[HID + col];
    const float bg = bias[2 * HID + col];
    const float bo = bias[3 * HID + col];
    float wi = 0.f, wf = 0.f, wg = 0.f, wo = 0.f;
    if (r1w) {
        wi = r1w[col]; wf = r1w[HID + col]; wg = r1w[2 * HID + col]; wo = r1w[3 * HID + col];
    }
#pragma unroll
    for (int mt = 0; mt < 2; ++mt) {
#pragma unroll
        for (int r = 0; r < 8; ++r) {
            const int row = m0 + mt * 16 + r + ((lane & 16) ? 8 : 0);
            const float xr = r1x ? r1x[row] : 0.0f;
            const float gi = acc[mt][0][r] + bi + xr * wi;
            const float gf = acc[mt][1][r] + bf + xr * wf;
            const float gg = acc[mt][2][r] + bg + xr * wg;
            const float go = acc[mt][3][r] + bo + xr * wo;
            const size_t idx = (size_t)row * HID + col;
            const float cn = sigm(gf) * C[idx] + sigm(gi) * tanh_fast(gg);
            const float hn = sigm(go) * tanh_fast(cn);
            C[idx] = cn;
            Hout[idx] = f2bf(hn);
        }
    }
}

// FC head: hid = relu(h1 @ W1^T + b1) [B,256];  pred = relu(hid @ W2^T + b2) [B]
__global__ __launch_bounds__(256, 1)
void fc_head_kernel(const unsigned short* __restrict__ H1,   // [B,H] bf16
                    const unsigned short* __restrict__ W1,   // [256,H] bf16
                    const float* __restrict__ b1,            // [256]
                    const float* __restrict__ W2,            // [256] f32
                    const float* __restrict__ b2,            // [1]
                    float* __restrict__ Ybuf,                // [B] next decoder input
                    float* __restrict__ Out, int t)          // Out[b*T_OUT + t]
{
    __shared__ float hid[64 * 256];
    const int lane = threadIdx.x & 31;
    const int wave = threadIdx.x >> 5;
    const int wm = wave & 1;
    const int wn = wave >> 1;                       // 4 groups x 64 cols = 256
    const int m0 = blockIdx.x * 64 + wm * 32;

    const v8f vz = {0.f, 0.f, 0.f, 0.f, 0.f, 0.f, 0.f, 0.f};
    v8f acc[2][4];
#pragma unroll
    for (int mt = 0; mt < 2; ++mt)
#pragma unroll
        for (int nt = 0; nt < 4; ++nt) acc[mt][nt] = vz;

    for (int k = 0; k < HID; k += 32) {
        v16bf a0 = load_frag(H1, m0,      HID, k);
        v16bf a1 = load_frag(H1, m0 + 16, HID, k);
#pragma unroll
        for (int nt = 0; nt < 4; ++nt) {
            v16bf b = load_frag(W1, wn * 64 + nt * 16, HID, k);
            acc[0][nt] = __builtin_amdgcn_wmma_f32_16x16x32_bf16(false, a0, false, b,
                                                                 (short)0, acc[0][nt], false, false);
            acc[1][nt] = __builtin_amdgcn_wmma_f32_16x16x32_bf16(false, a1, false, b,
                                                                 (short)0, acc[1][nt], false, false);
        }
    }
#pragma unroll
    for (int mt = 0; mt < 2; ++mt)
#pragma unroll
        for (int nt = 0; nt < 4; ++nt)
#pragma unroll
            for (int r = 0; r < 8; ++r) {
                const int rl = wm * 32 + mt * 16 + r + ((lane & 16) ? 8 : 0);
                const int cl = wn * 64 + nt * 16 + (lane & 15);
                hid[rl * 256 + cl] = fmaxf(acc[mt][nt][r] + b1[cl], 0.0f);
            }
    __syncthreads();

    if (threadIdx.x < 64) {
        const int row = blockIdx.x * 64 + threadIdx.x;
        float s = b2[0];
        for (int j = 0; j < 256; ++j) s += hid[threadIdx.x * 256 + j] * W2[j];
        s = fmaxf(s, 0.0f);
        Ybuf[row] = s;
        Out[(size_t)row * T_OUT + t] = s;
    }
}

__global__ void cvt_f32_bf16(const float* __restrict__ s, unsigned short* __restrict__ d, int n) {
    for (int i = blockIdx.x * blockDim.x + threadIdx.x; i < n; i += gridDim.x * blockDim.x)
        d[i] = f2bf(s[i]);
}
__global__ void zero_u32(unsigned int* __restrict__ p, int n) {
    for (int i = blockIdx.x * blockDim.x + threadIdx.x; i < n; i += gridDim.x * blockDim.x)
        p[i] = 0u;
}

static inline dim3 cvt_grid(int n) {
    int b = (n + 255) / 256;
    if (b > 4096) b = 4096;
    return dim3(b);
}

extern "C" void kernel_launch(void* const* d_in, const int* in_sizes, int n_in,
                              void* d_out, int out_size, void* d_ws, size_t ws_size,
                              hipStream_t stream) {
    (void)in_sizes; (void)n_in; (void)out_size; (void)ws_size;
    const float* x        = (const float*)d_in[0];
    const float* enc_Wih0 = (const float*)d_in[1];
    const float* enc_Whh0 = (const float*)d_in[2];
    const float* enc_b0   = (const float*)d_in[3];
    const float* enc_Wih1 = (const float*)d_in[4];
    const float* enc_Whh1 = (const float*)d_in[5];
    const float* enc_b1   = (const float*)d_in[6];
    const float* dec_Wih0 = (const float*)d_in[7];   // [2048,1] -> rank-1 fp32
    const float* dec_Whh0 = (const float*)d_in[8];
    const float* dec_b0   = (const float*)d_in[9];
    const float* dec_Wih1 = (const float*)d_in[10];
    const float* dec_Whh1 = (const float*)d_in[11];
    const float* dec_b1   = (const float*)d_in[12];
    const float* fc_W1    = (const float*)d_in[13];
    const float* fc_b1    = (const float*)d_in[14];
    const float* fc_W2    = (const float*)d_in[15];
    const float* fc_b2    = (const float*)d_in[16];
    float* out = (float*)d_out;

    char* ws = (char*)d_ws;
    size_t off = 0;
    auto carve = [&](size_t bytes) -> char* {
        char* p = ws + off;
        off += (bytes + 255) & ~(size_t)255;
        return p;
    };

    unsigned short* xbf   = (unsigned short*)carve((size_t)BATCH * T_IN * FEAT * 2);
    unsigned short* wih0e = (unsigned short*)carve((size_t)GATES * FEAT * 2);
    unsigned short* whh0e = (unsigned short*)carve((size_t)GATES * HID * 2);
    unsigned short* wih1e = (unsigned short*)carve((size_t)GATES * HID * 2);
    unsigned short* whh1e = (unsigned short*)carve((size_t)GATES * HID * 2);
    unsigned short* whh0d = (unsigned short*)carve((size_t)GATES * HID * 2);
    unsigned short* wih1d = (unsigned short*)carve((size_t)GATES * HID * 2);
    unsigned short* whh1d = (unsigned short*)carve((size_t)GATES * HID * 2);
    unsigned short* w1bf  = (unsigned short*)carve((size_t)256 * HID * 2);
    char* state_begin = ws + off;
    unsigned short* h0a = (unsigned short*)carve((size_t)BATCH * HID * 2);
    unsigned short* h0b = (unsigned short*)carve((size_t)BATCH * HID * 2);
    unsigned short* h1a = (unsigned short*)carve((size_t)BATCH * HID * 2);
    unsigned short* h1b = (unsigned short*)carve((size_t)BATCH * HID * 2);
    float* c0   = (float*)carve((size_t)BATCH * HID * 4);
    float* c1   = (float*)carve((size_t)BATCH * HID * 4);
    float* ybuf = (float*)carve((size_t)BATCH * 4);
    char* state_end = ws + off;

    // Zero all recurrent state (h, c, y) every call — deterministic.
    {
        int n = (int)((state_end - state_begin) / 4);
        zero_u32<<<cvt_grid(n), 256, 0, stream>>>((unsigned int*)state_begin, n);
    }
    // fp32 -> bf16 conversions
    cvt_f32_bf16<<<cvt_grid(BATCH * T_IN * FEAT), 256, 0, stream>>>(x, xbf, BATCH * T_IN * FEAT);
    cvt_f32_bf16<<<cvt_grid(GATES * FEAT), 256, 0, stream>>>(enc_Wih0, wih0e, GATES * FEAT);
    cvt_f32_bf16<<<cvt_grid(GATES * HID), 256, 0, stream>>>(enc_Whh0, whh0e, GATES * HID);
    cvt_f32_bf16<<<cvt_grid(GATES * HID), 256, 0, stream>>>(enc_Wih1, wih1e, GATES * HID);
    cvt_f32_bf16<<<cvt_grid(GATES * HID), 256, 0, stream>>>(enc_Whh1, whh1e, GATES * HID);
    cvt_f32_bf16<<<cvt_grid(GATES * HID), 256, 0, stream>>>(dec_Whh0, whh0d, GATES * HID);
    cvt_f32_bf16<<<cvt_grid(GATES * HID), 256, 0, stream>>>(dec_Wih1, wih1d, GATES * HID);
    cvt_f32_bf16<<<cvt_grid(GATES * HID), 256, 0, stream>>>(dec_Whh1, whh1d, GATES * HID);
    cvt_f32_bf16<<<cvt_grid(256 * HID), 256, 0, stream>>>(fc_W1, w1bf, 256 * HID);

    const dim3 cellGrid(BATCH / 64, HID / 64);   // 16 x 8
    const dim3 cellBlk(256);

    const unsigned short* h0p = h0a; unsigned short* h0n = h0b;
    const unsigned short* h1p = h1a; unsigned short* h1n = h1b;

    // Encoder: both layers in lockstep per timestep.
    for (int t = 0; t < T_IN; ++t) {
        lstm_cell_kernel<<<cellGrid, cellBlk, 0, stream>>>(
            h0p, xbf + (size_t)t * FEAT, FEAT, T_IN * FEAT,
            whh0e, wih0e, nullptr, nullptr, enc_b0, c0, h0n);
        lstm_cell_kernel<<<cellGrid, cellBlk, 0, stream>>>(
            h1p, h0n, HID, HID,
            whh1e, wih1e, nullptr, nullptr, enc_b1, c1, h1n);
        { const unsigned short* tp = h0p; h0p = h0n; h0n = (unsigned short*)tp; }
        { const unsigned short* tp = h1p; h1p = h1n; h1n = (unsigned short*)tp; }
    }

    // Decoder: autoregressive, scalar input ybuf (starts at 0).
    for (int t = 0; t < T_OUT; ++t) {
        lstm_cell_kernel<<<cellGrid, cellBlk, 0, stream>>>(
            h0p, nullptr, 0, 0,
            whh0d, nullptr, ybuf, dec_Wih0, dec_b0, c0, h0n);
        lstm_cell_kernel<<<cellGrid, cellBlk, 0, stream>>>(
            h1p, h0n, HID, HID,
            whh1d, wih1d, nullptr, nullptr, dec_b1, c1, h1n);
        fc_head_kernel<<<dim3(BATCH / 64), cellBlk, 0, stream>>>(
            h1n, w1bf, fc_b1, fc_W2, fc_b2, ybuf, out, t);
        { const unsigned short* tp = h0p; h0p = h0n; h0n = (unsigned short*)tp; }
        { const unsigned short* tp = h1p; h1p = h1n; h1n = (unsigned short*)tp; }
    }
}
